// ContinuousThoughtMachine_56659208569017
// MI455X (gfx1250) — matrix operand order repaired and checked by reading, not compiled, not run on hardware
//
#include <hip/hip_runtime.h>
#include <hip/hip_bf16.h>
#include <math.h>

// ---------------------------------------------------------------------------
// Continuous Thought Machine forward pass for MI455X (gfx1250, wave32).
// Matrix cores: V_WMMA_F32_16X16X32_F16 (f16 inputs, f32 accumulate).
// Dominant GEMM: conv2 as implicit im2col GEMM (32768 x 512 x 4608).
// The per-iteration skinny GEMMs (M=32, K=32896) are HBM-bound on q_w/outp_w,
// re-read 8x: all GEMM weights are converted+transposed to f16 [N][K] once per
// launch, halving the repeated weight traffic and making both LDS stages pure
// b128 load/store (no per-iteration cvt VALU in the hot loop).
// ---------------------------------------------------------------------------

typedef __attribute__((ext_vector_type(16))) _Float16 v16h;
typedef __attribute__((ext_vector_type(8)))  float    v8f;
typedef _Float16 f16;

#define BATCH   32
#define DIN     512
#define DMODEL  1024
#define MEMLEN  16
#define HEADS   8
#define HD      64
#define NSYNCH  256
#define REPRN   32896          // 256*257/2
#define MH      16
#define OUTD    256
#define NITERS  8
#define SLEN    256            // 16*16 tokens after pooling

struct IU { unsigned char i, j; };

// ---------------------------------------------------------------------------
// Weight prep: f32 [K,N] -> f16 [N,K] via LDS-tiled transpose (coalesced both
// sides). K, N multiples of 32 for every weight in this model.
// ---------------------------------------------------------------------------
__global__ __launch_bounds__(256)
void wprep_kernel(const float* __restrict__ in, f16* __restrict__ out, int K, int N)
{
    __shared__ f16 t[32][33];
    const int k0 = blockIdx.x * 32;
    const int n0 = blockIdx.y * 32;
    const int tx = threadIdx.x & 31;
    const int ty = threadIdx.x >> 5;     // 0..7
#pragma unroll
    for (int r = 0; r < 32; r += 8)
        t[ty + r][tx] = (f16)in[(size_t)(k0 + ty + r) * N + n0 + tx];
    __syncthreads();
#pragma unroll
    for (int r = 0; r < 32; r += 8)
        out[(size_t)(n0 + ty + r) * K + k0 + tx] = t[tx][ty + r];
}

// ---------------------------------------------------------------------------
// WMMA GEMM: C[M,N] = A[M,K] * Bt[N,K]^T (+ epilogue). Block = 128 thr = 4 waves.
// Tile: 32(M) x 64(N), K-step 32. Wave w computes N-subtile w (16 cols),
// M-subtiles 0 and 1 (two v8f accumulators).
// ---------------------------------------------------------------------------
enum { A_F16 = 0, A_F32 = 1, A_IM2COL = 2 };
enum { EPI_BIAS = 0, EPI_BNRELU = 1 };

template<int AMODE, int EPI>
__global__ __launch_bounds__(128)
void gemm_wmma_kernel(const void* __restrict__ Aany, int lda,
                      const f16* __restrict__ Bt,           // [N][K] f16
                      const float* __restrict__ bias,
                      const float* __restrict__ bns, const float* __restrict__ bnb,
                      const float* __restrict__ bnm, const float* __restrict__ bnv,
                      void* __restrict__ Cany, int N, int K)
{
    // padded strides (40 halves = 80B, multiple of 16B) keep b128 LDS ops aligned
    __shared__ f16 As[32 * 40];
    __shared__ f16 Bs[64 * 40];

    const int tid  = threadIdx.x;
    const int lane = tid & 31;
    const int wave = tid >> 5;
    const int m0   = blockIdx.y * 32;
    const int n0   = blockIdx.x * 64;

    v8f acc0, acc1;
#pragma unroll
    for (int i = 0; i < 8; ++i) { acc0[i] = 0.f; acc1[i] = 0.f; }

    const int arow = tid >> 2;        // 0..31 : M row within tile
    const int aseg = (tid & 3) * 8;   // K offset (8 halves per thread)
    const int bn   = tid >> 1;        // 0..63 : N row of Bt within tile
    const int bseg = (tid & 1) * 16;  // K offset (16 halves per thread)

    for (int k0 = 0; k0 < K; k0 += 32) {
        __syncthreads();
        // ---- stage A tile (32 rows x 32 halves) ----
        {
            union H8 { f16 h[8]; float4 q; } pk;
            if (AMODE == A_F16) {
                const f16* A = (const f16*)Aany;
                pk.q = *(const float4*)(A + (size_t)(m0 + arow) * lda + k0 + aseg);
            } else if (AMODE == A_F32) {
                const float* A = (const float*)Aany;
                const float4 lo = *(const float4*)(A + (size_t)(m0 + arow) * lda + k0 + aseg);
                const float4 hi = *(const float4*)(A + (size_t)(m0 + arow) * lda + k0 + aseg + 4);
                pk.h[0]=(f16)lo.x; pk.h[1]=(f16)lo.y; pk.h[2]=(f16)lo.z; pk.h[3]=(f16)lo.w;
                pk.h[4]=(f16)hi.x; pk.h[5]=(f16)hi.y; pk.h[6]=(f16)hi.z; pk.h[7]=(f16)hi.w;
            } else { // A_IM2COL: conv2 on h1 [B,32,32,512] (f16), 3x3 SAME
                const f16* H1 = (const f16*)Aany;
                const int m = m0 + arow;
                const int b = m >> 10;
                const int y = (m >> 5) & 31;
                const int x = m & 31;
                const int patch = k0 >> 9;              // K chunk of 32 stays in one (ky,kx)
                const int ky = patch / 3, kx = patch - ky * 3;
                const int ci = (k0 & 511) + aseg;
                const int yy = y + ky - 1, xx = x + kx - 1;
                if ((unsigned)yy < 32u && (unsigned)xx < 32u) {
                    pk.q = *(const float4*)(H1 + ((((size_t)b * 32 + yy) * 32 + xx) * 512 + ci));
                } else {
#pragma unroll
                    for (int i = 0; i < 8; ++i) pk.h[i] = (f16)0.f;
                }
            }
            *(float4*)(&As[arow * 40 + aseg]) = pk.q;
        }
        // ---- stage B tile: Bt rows are K-contiguous f16 -> pure b128 traffic ----
        {
            const f16* src = Bt + (size_t)(n0 + bn) * K + k0 + bseg;
            const float4 q0 = *(const float4*)(src);
            const float4 q1 = *(const float4*)(src + 8);
            *(float4*)(&Bs[bn * 40 + bseg])     = q0;
            *(float4*)(&Bs[bn * 40 + bseg + 8]) = q1;
            if (k0 + 32 < K)
                __builtin_prefetch(src + 32, 0, 0);     // global_prefetch_b8
        }
        __syncthreads();

        // ---- gather fragments per ISA 7.12.2 layouts, 2 WMMAs ----
        union AF { v16h v; float4 q[2]; } fa0, fa1, fb;
        const int hsel = lane >> 4;     // lane half selects K sub-range
        const int mrow = lane & 15;
        {
            // A 16x32 f16: lane m = lane&15; halves 0-7 -> K=base8+0..7,
            //              halves 8-15 -> K=base8+16..23 (base8 = 8*(lane>=16))
            const f16* p0 = &As[mrow * 40 + hsel * 8];
            fa0.q[0] = *(const float4*)(p0);
            fa0.q[1] = *(const float4*)(p0 + 16);
            const f16* p1 = &As[(16 + mrow) * 40 + hsel * 8];
            fa1.q[0] = *(const float4*)(p1);
            fa1.q[1] = *(const float4*)(p1 + 16);
            // B 32x16 f16: lane n = lane&15; same K order as A (base8+0..7, base8+16..23)
            const f16* pb = &Bs[(wave * 16 + mrow) * 40 + hsel * 8];
            fb.q[0] = *(const float4*)(pb);
            fb.q[1] = *(const float4*)(pb + 16);
        }
        acc0 = __builtin_amdgcn_wmma_f32_16x16x32_f16(false, fa0.v, false, fb.v,
                                                      (short)0, acc0, false, false);
        acc1 = __builtin_amdgcn_wmma_f32_16x16x32_f16(false, fa1.v, false, fb.v,
                                                      (short)0, acc1, false, false);
    }

    // ---- epilogue: C/D 16x16 f32 layout: N = lane&15, M = r + 8*(lane>=16) ----
    const int n    = n0 + wave * 16 + (lane & 15);
    const int mofs = 8 * (lane >> 4);
#pragma unroll
    for (int r = 0; r < 8; ++r) {
        const int mA = m0 + mofs + r;
        const int mB = m0 + 16 + mofs + r;
        if (EPI == EPI_BIAS) {
            float* C = (float*)Cany;
            C[(size_t)mA * N + n] = acc0[r] + bias[n];
            C[(size_t)mB * N + n] = acc1[r] + bias[n];
        } else { // conv2 bias + BN + ReLU -> f16
            f16* C = (f16*)Cany;
            const float inv = rsqrtf(bnv[n] + 1e-5f) * bns[n];
            float t0 = (acc0[r] + bias[n] - bnm[n]) * inv + bnb[n];
            float t1 = (acc1[r] + bias[n] - bnm[n]) * inv + bnb[n];
            C[(size_t)mA * N + n] = (f16)fmaxf(t0, 0.f);
            C[(size_t)mB * N + n] = (f16)fmaxf(t1, 0.f);
        }
    }
}

// ---------------------------------------------------------------------------
// conv1 (3->512, 3x3 SAME) + BN + ReLU + 2x2 maxpool fused. K=27 is too small
// for WMMA; VALU with the 4x4x3 input patch staged in LDS (shared across 512 c).
// ---------------------------------------------------------------------------
__global__ __launch_bounds__(512)
void conv1_kernel(const float* __restrict__ x,
                  const float* __restrict__ w, const float* __restrict__ bconv,
                  const float* __restrict__ s, const float* __restrict__ bb,
                  const float* __restrict__ m, const float* __restrict__ v,
                  f16* __restrict__ h1)
{
    __shared__ float patch[4][4][3];
    const int bid = blockIdx.x;                 // (b, yo, xo) over 32x32x32
    const int xo = bid & 31;
    const int yo = (bid >> 5) & 31;
    const int bi = bid >> 10;
    const int tid = threadIdx.x;
    if (tid < 48) {
        const int ci = tid % 3;
        const int px = (tid / 3) & 3;
        const int py = tid / 12;
        const int yy = yo * 2 - 1 + py;
        const int xx = xo * 2 - 1 + px;
        float val = 0.f;
        if ((unsigned)yy < 64u && (unsigned)xx < 64u)
            val = x[(((size_t)bi * 64 + yy) * 64 + xx) * 3 + ci];
        patch[py][px][ci] = val;
    }
    __syncthreads();
    const int c = tid;                          // 512 channels
    const float inv = rsqrtf(v[c] + 1e-5f) * s[c];
    float best = -1e30f;
#pragma unroll
    for (int py = 0; py < 2; ++py)
#pragma unroll
    for (int px = 0; px < 2; ++px) {
        float acc = bconv[c];
#pragma unroll
        for (int ky = 0; ky < 3; ++ky)
#pragma unroll
        for (int kx = 0; kx < 3; ++kx)
#pragma unroll
        for (int ci = 0; ci < 3; ++ci)
            acc += patch[py + ky][px + kx][ci] * w[(((ky * 3 + kx) * 3 + ci) << 9) + c];
        best = fmaxf(best, fmaxf((acc - m[c]) * inv + bb[c], 0.f));
    }
    h1[(((size_t)bi * 32 + yo) * 32 + xo) * 512 + c] = (f16)best;
}

__global__ void pool2_kernel(const f16* __restrict__ h2, f16* __restrict__ hp)
{
    const int idx = blockIdx.x * blockDim.x + threadIdx.x;  // B*16*16*512
    if (idx >= BATCH * 16 * 16 * 512) return;
    const int c  = idx & 511;
    const int xo = (idx >> 9) & 15;
    const int yo = (idx >> 13) & 15;
    const int b  = idx >> 17;
    float best = -1e30f;
#pragma unroll
    for (int dy = 0; dy < 2; ++dy)
#pragma unroll
    for (int dx = 0; dx < 2; ++dx) {
        float t = (float)h2[(((size_t)b * 32 + yo * 2 + dy) * 32 + xo * 2 + dx) * 512 + c];
        best = fmaxf(best, t);
    }
    hp[idx] = (f16)best;
}

// LayerNorm rows of f32 [R,C] -> f16
__global__ __launch_bounds__(256)
void ln_f16_kernel(const float* __restrict__ in, const float* __restrict__ s,
                   const float* __restrict__ b, f16* __restrict__ out, int C)
{
    __shared__ float red[256];
    const int row = blockIdx.x;
    const float* r = in + (size_t)row * C;
    float sum = 0.f, sq = 0.f;
    for (int j = threadIdx.x; j < C; j += 256) { float t = r[j]; sum += t; sq += t * t; }
    red[threadIdx.x] = sum; __syncthreads();
    for (int o = 128; o > 0; o >>= 1) { if (threadIdx.x < o) red[threadIdx.x] += red[threadIdx.x + o]; __syncthreads(); }
    const float mean = red[0] / C; __syncthreads();
    red[threadIdx.x] = sq; __syncthreads();
    for (int o = 128; o > 0; o >>= 1) { if (threadIdx.x < o) red[threadIdx.x] += red[threadIdx.x + o]; __syncthreads(); }
    const float inv = rsqrtf(red[0] / C - mean * mean + 1e-6f);
    for (int j = threadIdx.x; j < C; j += 256)
        out[(size_t)row * C + j] = (f16)((r[j] - mean) * inv * s[j] + b[j]);
}

// triu index table for N_SYNCH=256
__global__ void iu_kernel(IU* tab)
{
    const int i = blockIdx.x;     // 0..255
    const int j = threadIdx.x;    // 0..255
    if (j < i) return;
    const int off = i * NSYNCH - (i * (i - 1)) / 2;
    tab[off + (j - i)] = IU{(unsigned char)i, (unsigned char)j};
}

__global__ void init_state_kernel(const float* __restrict__ sa, const float* __restrict__ st,
                                  float* __restrict__ act, float* __restrict__ trace)
{
    const int idx = blockIdx.x * blockDim.x + threadIdx.x;   // B*DMODEL
    if (idx >= BATCH * DMODEL) return;
    const int d = idx & (DMODEL - 1);
    act[idx] = sa[d];
#pragma unroll
    for (int mm = 0; mm < MEMLEN; ++mm)
        trace[(size_t)idx * MEMLEN + mm] = st[d * MEMLEN + mm];
}

// synchronization representation: s = alpha*rsqrt(beta) with EMA recurrence
__global__ void sync_kernel(const float* __restrict__ act, int off,
                            float* __restrict__ alpha, float* __restrict__ beta,
                            const float* __restrict__ decay, int init,
                            const IU* __restrict__ tab, f16* __restrict__ sout)
{
    const int idx = blockIdx.x * blockDim.x + threadIdx.x;   // B*REPRN
    if (idx >= BATCH * REPRN) return;
    const int b  = idx / REPRN;
    const int rr = idx - b * REPRN;
    const IU p = tab[rr];
    const float* a = act + (size_t)b * DMODEL + off;
    const float pp = a[p.i] * a[p.j];
    float al, be;
    if (init) { al = pp; be = 1.f; }
    else {
        const float r = __expf(-fminf(fmaxf(decay[rr], 0.f), 15.f));
        al = r * alpha[idx] + pp;
        be = r * beta[idx] + 1.f;
    }
    alpha[idx] = al; beta[idx] = be;
    sout[idx] = (f16)(al * rsqrtf(be));
}

// per-(b,h) attention: logits over 256 tokens, softmax, weighted sum of v
__global__ __launch_bounds__(256)
void attn_kernel(const float* __restrict__ q, const float* __restrict__ k,
                 const float* __restrict__ v, float* __restrict__ ctx)
{
    __shared__ float qs[HD];
    __shared__ float lg[SLEN];
    __shared__ float red[256];
    const int h = blockIdx.x & (HEADS - 1);
    const int b = blockIdx.x >> 3;
    const int tid = threadIdx.x;
    if (tid < HD) qs[tid] = q[(size_t)b * DIN + h * HD + tid];
    __syncthreads();
    {
        const float* kr = k + ((size_t)(b * SLEN + tid) * DIN + h * HD);
        float acc = 0.f;
#pragma unroll 8
        for (int d = 0; d < HD; ++d) acc += qs[d] * kr[d];
        lg[tid] = acc * 0.125f;                 // / sqrt(64)
    }
    __syncthreads();
    red[tid] = lg[tid]; __syncthreads();
    for (int o = 128; o > 0; o >>= 1) { if (tid < o) red[tid] = fmaxf(red[tid], red[tid + o]); __syncthreads(); }
    const float mx = red[0]; __syncthreads();
    const float e = __expf(lg[tid] - mx);
    red[tid] = e; __syncthreads();
    for (int o = 128; o > 0; o >>= 1) { if (tid < o) red[tid] += red[tid + o]; __syncthreads(); }
    const float inv = 1.f / red[0];
    __syncthreads();
    lg[tid] = e * inv;
    __syncthreads();
    if (tid < HD) {
        float acc = 0.f;
        const float* vb = v + ((size_t)b * SLEN) * DIN + h * HD + tid;
        for (int s = 0; s < SLEN; ++s) acc += lg[s] * vb[(size_t)s * DIN];
        ctx[(size_t)b * DIN + h * HD + tid] = acc;
    }
}

__global__ void cat_kernel(const float* __restrict__ attn, const float* __restrict__ act,
                           f16* __restrict__ out)
{
    const int idx = blockIdx.x * blockDim.x + threadIdx.x;   // B*1536
    if (idx >= BATCH * (DIN + DMODEL)) return;
    const int b = idx / (DIN + DMODEL), j = idx - b * (DIN + DMODEL);
    const float t = (j < DIN) ? attn[b * DIN + j] : act[b * DMODEL + (j - DIN)];
    out[idx] = (f16)t;
}

// GLU -> LayerNorm -> trace shift+append
__global__ __launch_bounds__(256)
void glu_ln_trace_kernel(const float* __restrict__ y,
                         const float* __restrict__ s, const float* __restrict__ bb,
                         float* __restrict__ trace)
{
    __shared__ float st[DMODEL];
    __shared__ float red[256];
    const int b = blockIdx.x;
    const int tid = threadIdx.x;
    float sum = 0.f, sq = 0.f;
    for (int j = tid; j < DMODEL; j += 256) {
        const float a = y[(size_t)b * 2 * DMODEL + j];
        const float g = y[(size_t)b * 2 * DMODEL + DMODEL + j];
        const float t = a * (1.f / (1.f + __expf(-g)));
        st[j] = t; sum += t; sq += t * t;
    }
    red[tid] = sum; __syncthreads();
    for (int o = 128; o > 0; o >>= 1) { if (tid < o) red[tid] += red[tid + o]; __syncthreads(); }
    const float mean = red[0] / DMODEL; __syncthreads();
    red[tid] = sq; __syncthreads();
    for (int o = 128; o > 0; o >>= 1) { if (tid < o) red[tid] += red[tid + o]; __syncthreads(); }
    const float inv = rsqrtf(red[0] / DMODEL - mean * mean + 1e-6f);
    __syncthreads();
    for (int j = tid; j < DMODEL; j += 256) {
        const float sv = (st[j] - mean) * inv * s[j] + bb[j];
        float* tr = trace + ((size_t)b * DMODEL + j) * MEMLEN;
#pragma unroll
        for (int mm = 0; mm < MEMLEN - 1; ++mm) tr[mm] = tr[mm + 1];
        tr[MEMLEN - 1] = sv;
    }
}

// per-neuron MLP: trace[16] -> 32 (GLU -> 16) -> 2 -> act = a*sigmoid(g)
__global__ void neuron_kernel(const float* __restrict__ trace,
                              const float* __restrict__ w1, const float* __restrict__ b1,
                              const float* __restrict__ w2, const float* __restrict__ b2,
                              float* __restrict__ act)
{
    const int idx = blockIdx.x * blockDim.x + threadIdx.x;   // B*DMODEL
    if (idx >= BATCH * DMODEL) return;
    const int d = idx & (DMODEL - 1);
    float tr[MEMLEN];
    const float* t = trace + (size_t)idx * MEMLEN;
#pragma unroll
    for (int mm = 0; mm < MEMLEN; ++mm) tr[mm] = t[mm];
    float hb[2 * MH];
#pragma unroll
    for (int j = 0; j < 2 * MH; ++j) {
        float acc = b1[d * 2 * MH + j];
#pragma unroll
        for (int mm = 0; mm < MEMLEN; ++mm)
            acc += tr[mm] * w1[((size_t)mm * 2 * MH + j) * DMODEL + d];
        hb[j] = acc;
    }
    float tv[MH];
#pragma unroll
    for (int j = 0; j < MH; ++j)
        tv[j] = hb[j] * (1.f / (1.f + __expf(-hb[MH + j])));
    float o0 = b2[d * 2 + 0], o1 = b2[d * 2 + 1];
#pragma unroll
    for (int j = 0; j < MH; ++j) {
        o0 += tv[j] * w2[((size_t)j * 2 + 0) * DMODEL + d];
        o1 += tv[j] * w2[((size_t)j * 2 + 1) * DMODEL + d];
    }
    act[idx] = o0 * (1.f / (1.f + __expf(-o1)));
}

// log-softmax entropy + scatter into output (preds [B,256,8], certs [B,2,8])
__global__ __launch_bounds__(256)
void pred_kernel(const float* __restrict__ pred, float* __restrict__ out, int it)
{
    __shared__ float red[256];
    const int b = blockIdx.x;
    const int tid = threadIdx.x;
    const float x = pred[b * OUTD + tid];
    red[tid] = x; __syncthreads();
    for (int o = 128; o > 0; o >>= 1) { if (tid < o) red[tid] = fmaxf(red[tid], red[tid + o]); __syncthreads(); }
    const float mx = red[0]; __syncthreads();
    const float e = __expf(x - mx);
    red[tid] = e; __syncthreads();
    for (int o = 128; o > 0; o >>= 1) { if (tid < o) red[tid] += red[tid + o]; __syncthreads(); }
    const float lse = __logf(red[0]); __syncthreads();
    const float lp = x - mx - lse;
    red[tid] = __expf(lp) * lp; __syncthreads();
    for (int o = 128; o > 0; o >>= 1) { if (tid < o) red[tid] += red[tid + o]; __syncthreads(); }
    const float ne = -red[0] / logf(256.f);
    out[((size_t)b * OUTD + tid) * NITERS + it] = x;
    if (tid == 0) {
        float* certs = out + (size_t)BATCH * OUTD * NITERS;
        certs[(b * 2 + 0) * NITERS + it] = ne;
        certs[(b * 2 + 1) * NITERS + it] = 1.f - ne;
    }
}

// ---------------------------------------------------------------------------
extern "C" void kernel_launch(void* const* d_in, const int* in_sizes, int n_in,
                              void* d_out, int out_size, void* d_ws, size_t ws_size,
                              hipStream_t stream)
{
    (void)in_sizes; (void)n_in; (void)out_size; (void)ws_size;
    const float* x        = (const float*)d_in[0];
    const float* conv1_w  = (const float*)d_in[1];
    const float* conv1_b  = (const float*)d_in[2];
    const float* bn1_s    = (const float*)d_in[3];
    const float* bn1_b    = (const float*)d_in[4];
    const float* bn1_m    = (const float*)d_in[5];
    const float* bn1_v    = (const float*)d_in[6];
    const float* conv2_w  = (const float*)d_in[7];
    const float* conv2_b  = (const float*)d_in[8];
    const float* bn2_s    = (const float*)d_in[9];
    const float* bn2_b    = (const float*)d_in[10];
    const float* bn2_m    = (const float*)d_in[11];
    const float* bn2_v    = (const float*)d_in[12];
    const float* kv_w     = (const float*)d_in[13];
    const float* kv_b     = (const float*)d_in[14];
    const float* kvln_s   = (const float*)d_in[15];
    const float* kvln_b   = (const float*)d_in[16];
    const float* q_w      = (const float*)d_in[17];
    const float* q_b      = (const float*)d_in[18];
    const float* aq_w     = (const float*)d_in[19];
    const float* aq_b     = (const float*)d_in[20];
    const float* ak_w     = (const float*)d_in[21];
    const float* ak_b     = (const float*)d_in[22];
    const float* av_w     = (const float*)d_in[23];
    const float* av_b     = (const float*)d_in[24];
    const float* ao_w     = (const float*)d_in[25];
    const float* ao_b     = (const float*)d_in[26];
    const float* syn_w    = (const float*)d_in[27];
    const float* syn_b    = (const float*)d_in[28];
    const float* synln_s  = (const float*)d_in[29];
    const float* synln_b  = (const float*)d_in[30];
    const float* sl1_w    = (const float*)d_in[31];
    const float* sl1_b    = (const float*)d_in[32];
    const float* sl2_w    = (const float*)d_in[33];
    const float* sl2_b    = (const float*)d_in[34];
    const float* outp_w   = (const float*)d_in[35];
    const float* outp_b   = (const float*)d_in[36];
    const float* start_a  = (const float*)d_in[37];
    const float* start_t  = (const float*)d_in[38];
    const float* dec_act  = (const float*)d_in[39];
    const float* dec_out  = (const float*)d_in[40];

    char* ws = (char*)d_ws;
    // Region 0 (33.5MB): h1 (f16) during conv; kvtmp (f32) afterwards.
    // Region 1 (33.5MB): h2 (f16) during conv; k,v (f32) afterwards.
    const size_t R0 = 0, R1 = 33554432;
    f16*   h1    = (f16*)(ws + R0);
    f16*   h2    = (f16*)(ws + R1);
    float* kvtmp = (float*)(ws + R0);
    float* kbuf  = (float*)(ws + R1);
    float* vbuf  = (float*)(ws + R1 + 16777216);

    size_t off = 67108864;
    auto alloc = [&](size_t bytes) -> void* {
        void* p = (void*)(ws + off);
        off += (bytes + 255) & ~(size_t)255;
        return p;
    };
    f16*   hp     = (f16*)  alloc((size_t)BATCH * SLEN * DIN * 2);
    f16*   kvh    = (f16*)  alloc((size_t)BATCH * SLEN * DIN * 2);
    float* a_act  = (float*)alloc((size_t)BATCH * REPRN * 4);
    float* b_act  = (float*)alloc((size_t)BATCH * REPRN * 4);
    float* a_out  = (float*)alloc((size_t)BATCH * REPRN * 4);
    float* b_out  = (float*)alloc((size_t)BATCH * REPRN * 4);
    f16*   s_act  = (f16*)  alloc((size_t)BATCH * REPRN * 2);
    f16*   s_out  = (f16*)  alloc((size_t)BATCH * REPRN * 2);
    IU*    tab    = (IU*)   alloc((size_t)REPRN * sizeof(IU));
    float* act    = (float*)alloc((size_t)BATCH * DMODEL * 4);
    float* trace  = (float*)alloc((size_t)BATCH * DMODEL * MEMLEN * 4);
    float* q1     = (float*)alloc((size_t)BATCH * DIN * 4);
    float* qv     = (float*)alloc((size_t)BATCH * DIN * 4);
    float* ctx    = (float*)alloc((size_t)BATCH * DIN * 4);
    float* aoo    = (float*)alloc((size_t)BATCH * DIN * 4);
    f16*   catb   = (f16*)  alloc((size_t)BATCH * (DIN + DMODEL) * 2);
    float* ybuf   = (float*)alloc((size_t)BATCH * 2 * DMODEL * 4);
    float* predb  = (float*)alloc((size_t)BATCH * OUTD * 4);
    // f16 [N][K] transposed weights (one-time prep per launch)
    f16* conv2t = (f16*)alloc((size_t)4608 * 512 * 2);
    f16* kvt    = (f16*)alloc((size_t)512 * 512 * 2);
    f16* akt    = (f16*)alloc((size_t)512 * 512 * 2);
    f16* avt    = (f16*)alloc((size_t)512 * 512 * 2);
    f16* aqt    = (f16*)alloc((size_t)512 * 512 * 2);
    f16* aot    = (f16*)alloc((size_t)512 * 512 * 2);
    f16* qt     = (f16*)alloc((size_t)REPRN * DIN * 2);
    f16* synt   = (f16*)alloc((size_t)1536 * 2048 * 2);
    f16* outpt  = (f16*)alloc((size_t)REPRN * OUTD * 2);
    float* outp_f = (float*)d_out;

    // ---- one-time weight prep: f32 [K,N] -> f16 [N,K] ----
    wprep_kernel<<<dim3(4608 / 32, 512 / 32), 256, 0, stream>>>(conv2_w, conv2t, 4608, 512);
    wprep_kernel<<<dim3(512 / 32, 512 / 32), 256, 0, stream>>>(kv_w, kvt, 512, 512);
    wprep_kernel<<<dim3(512 / 32, 512 / 32), 256, 0, stream>>>(ak_w, akt, 512, 512);
    wprep_kernel<<<dim3(512 / 32, 512 / 32), 256, 0, stream>>>(av_w, avt, 512, 512);
    wprep_kernel<<<dim3(512 / 32, 512 / 32), 256, 0, stream>>>(aq_w, aqt, 512, 512);
    wprep_kernel<<<dim3(512 / 32, 512 / 32), 256, 0, stream>>>(ao_w, aot, 512, 512);
    wprep_kernel<<<dim3(REPRN / 32, DIN / 32), 256, 0, stream>>>(q_w, qt, REPRN, DIN);
    wprep_kernel<<<dim3(1536 / 32, 2048 / 32), 256, 0, stream>>>(syn_w, synt, 1536, 2048);
    wprep_kernel<<<dim3(REPRN / 32, OUTD / 32), 256, 0, stream>>>(outp_w, outpt, REPRN, OUTD);

    // ---- feature backbone ----
    conv1_kernel<<<32 * 32 * 32, 512, 0, stream>>>(x, conv1_w, conv1_b,
                                                   bn1_s, bn1_b, bn1_m, bn1_v, h1);
    gemm_wmma_kernel<A_IM2COL, EPI_BNRELU><<<dim3(512 / 64, 32768 / 32), 128, 0, stream>>>(
        h1, 0, conv2t, conv2_b, bn2_s, bn2_b, bn2_m, bn2_v, h2, 512, 4608);
    pool2_kernel<<<(BATCH * 16 * 16 * 512 + 255) / 256, 256, 0, stream>>>(h2, hp);

    gemm_wmma_kernel<A_F16, EPI_BIAS><<<dim3(512 / 64, 8192 / 32), 128, 0, stream>>>(
        hp, DIN, kvt, kv_b, nullptr, nullptr, nullptr, nullptr, kvtmp, DIN, DIN);
    ln_f16_kernel<<<BATCH * SLEN, 256, 0, stream>>>(kvtmp, kvln_s, kvln_b, kvh, DIN);
    gemm_wmma_kernel<A_F16, EPI_BIAS><<<dim3(512 / 64, 8192 / 32), 128, 0, stream>>>(
        kvh, DIN, akt, ak_b, nullptr, nullptr, nullptr, nullptr, kbuf, DIN, DIN);
    gemm_wmma_kernel<A_F16, EPI_BIAS><<<dim3(512 / 64, 8192 / 32), 128, 0, stream>>>(
        kvh, DIN, avt, av_b, nullptr, nullptr, nullptr, nullptr, vbuf, DIN, DIN);

    // ---- recurrent state init ----
    iu_kernel<<<256, 256, 0, stream>>>(tab);
    init_state_kernel<<<(BATCH * DMODEL + 255) / 256, 256, 0, stream>>>(start_a, start_t, act, trace);
    const int syncBlocks = (BATCH * REPRN) / 256;   // 4112 exact
    sync_kernel<<<syncBlocks, 256, 0, stream>>>(act, 0, a_out, b_out, dec_out, 1, tab, s_out);

    // ---- 8 thought iterations ----
    for (int it = 0; it < NITERS; ++it) {
        sync_kernel<<<syncBlocks, 256, 0, stream>>>(act, DMODEL - NSYNCH, a_act, b_act,
                                                    dec_act, it == 0 ? 1 : 0, tab, s_act);
        gemm_wmma_kernel<A_F16, EPI_BIAS><<<dim3(512 / 64, 1), 128, 0, stream>>>(
            s_act, REPRN, qt, q_b, nullptr, nullptr, nullptr, nullptr, q1, DIN, REPRN);
        gemm_wmma_kernel<A_F32, EPI_BIAS><<<dim3(512 / 64, 1), 128, 0, stream>>>(
            q1, DIN, aqt, aq_b, nullptr, nullptr, nullptr, nullptr, qv, DIN, DIN);
        attn_kernel<<<BATCH * HEADS, 256, 0, stream>>>(qv, kbuf, vbuf, ctx);
        gemm_wmma_kernel<A_F32, EPI_BIAS><<<dim3(512 / 64, 1), 128, 0, stream>>>(
            ctx, DIN, aot, ao_b, nullptr, nullptr, nullptr, nullptr, aoo, DIN, DIN);
        cat_kernel<<<(BATCH * (DIN + DMODEL) + 255) / 256, 256, 0, stream>>>(aoo, act, catb);
        gemm_wmma_kernel<A_F16, EPI_BIAS><<<dim3(2048 / 64, 1), 128, 0, stream>>>(
            catb, DIN + DMODEL, synt, syn_b, nullptr, nullptr, nullptr, nullptr,
            ybuf, 2 * DMODEL, DIN + DMODEL);
        glu_ln_trace_kernel<<<BATCH, 256, 0, stream>>>(ybuf, synln_s, synln_b, trace);
        neuron_kernel<<<(BATCH * DMODEL + 255) / 256, 256, 0, stream>>>(trace, sl1_w, sl1_b,
                                                                        sl2_w, sl2_b, act);
        sync_kernel<<<syncBlocks, 256, 0, stream>>>(act, 0, a_out, b_out, dec_out, 0, tab, s_out);
        gemm_wmma_kernel<A_F16, EPI_BIAS><<<dim3(OUTD / 64, 1), 128, 0, stream>>>(
            s_out, REPRN, outpt, outp_b, nullptr, nullptr, nullptr, nullptr,
            predb, OUTD, REPRN);
        pred_kernel<<<BATCH, 256, 0, stream>>>(predb, outp_f, it);
    }
}
